// CrossModalHyperMamba_60713657697151
// MI455X (gfx1250) — matrix operand order repaired and checked by reading, not compiled, not run on hardware
//
#include <hip/hip_runtime.h>
#include <math.h>

// ---------------------------------------------------------------------------
// CDNA5 / gfx1250 implementation of CrossModalHyperMamba forward.
// All 1x1-conv / linear ops -> f16 WMMA GEMM (v_wmma_f32_16x16x32_f16) with
// fused bias + activation.  Depthwise convs, causal conv1d, selective scan,
// softmax, RMSNorm, SE gates etc. are VALU kernels.
// ---------------------------------------------------------------------------

typedef __attribute__((ext_vector_type(16))) _Float16 v16h;
typedef __attribute__((ext_vector_type(8)))  float    v8f;

// problem constants
constexpr int kB   = 4;
constexpr int kC   = 128;
constexpr int kH   = 64;
constexpr int kW   = 64;
constexpr int kHW  = kH * kW;        // 4096 tokens per image
constexpr int kCh  = 64;             // c_ = C/2
constexpr int kQ   = 32;             // DSC3k hidden
constexpr int kE   = 8;              // NUM_HE
constexpr int kS   = 8;              // D_STATE
constexpr int kDT  = 8;              // DT_RANK

enum { ACT_NONE = 0, ACT_SILU = 1, ACT_SIG = 2, ACT_TANH = 3, ACT_GELU = 4, ACT_SOFTPLUS = 5 };

__device__ __forceinline__ float d_sigmoid(float v) { return 1.f / (1.f + __expf(-v)); }
__device__ __forceinline__ float d_silu(float v)    { return v * d_sigmoid(v); }
__device__ __forceinline__ float d_gelu(float v)    { return 0.5f * v * (1.f + erff(v * 0.70710678118654752f)); }
__device__ __forceinline__ float d_softplus(float v){ return (v > 20.f) ? v : log1pf(__expf(v)); }
__device__ __forceinline__ float d_act(float v, int a) {
  switch (a) {
    case ACT_SILU:     return d_silu(v);
    case ACT_SIG:      return d_sigmoid(v);
    case ACT_TANH:     return tanhf(v);
    case ACT_GELU:     return d_gelu(v);
    case ACT_SOFTPLUS: return d_softplus(v);
  }
  return v;
}

// scan-direction permutation: scan position t reads original token perm(t)
__device__ __forceinline__ int perm_idx(int dir, int t, int Hh, int Wd) {
  int L = Hh * Wd;
  if (dir == 0) return t;
  if (dir == 1) return L - 1 - t;
  int j = (dir == 2) ? t : (L - 1 - t);
  return (j % Hh) * Wd + (j / Hh);
}

// ---------------------------------------------------------------------------
// WMMA GEMM:  Out[b] = act(alpha * A @ B[b] + bias)
//   A  : row-major [M,K], lda=K, batch stride sA (0 for shared weights)
//   B  : row-major [K,N], leading dim ldb, batch stride sB
//   Out: row stride ldo, batch stride sO
// Block = 256 threads (8 wave32 waves). Block tile 32(M) x 128(N); each wave
// owns two 16x16 D tiles stacked in M (shared B fragment -> 2 v_wmma / step).
// K stepped by 32 through LDS with fp32->f16 conversion.  Fast path stages
// with float4 (global_load_b128); guarded scalar path only for edge tiles.
// ---------------------------------------------------------------------------
__global__ __launch_bounds__(256) void wmma_gemm_bias_act(
    const float* __restrict__ A, long sA,
    const float* __restrict__ Bm, int ldb, long sB,
    const float* __restrict__ bias,
    float* __restrict__ Out, int ldo, long sO,
    int M, int N, int K, float alpha, int act)
{
  __shared__ _Float16 As[32][40];    // [m][k]
  __shared__ _Float16 Bs[128][40];   // [n][k]  (k-contiguous per column)

  const int b = blockIdx.z;
  const float* Ab = A  + (long)b * sA;
  const float* Bb = Bm + (long)b * sB;
  float*       Ob = Out + (long)b * sO;

  const int t      = threadIdx.x;
  const int wid    = t >> 5;
  const int lane   = t & 31;
  const int row16  = lane & 15;
  const int khalf  = lane >> 4;
  const int mbase  = blockIdx.y * 32;
  const int nblock = blockIdx.x * 128;
  const int coln   = (wid << 4) + row16;   // local N column for this lane

  v8f acc0 = {}, acc1 = {};

  const bool fullM = (mbase + 32 <= M);
  const bool fullN = (nblock + 128 <= N) && ((ldb & 3) == 0);
  const bool k4    = ((K & 3) == 0);

  for (int k0 = 0; k0 < K; k0 += 32) {
    const bool fullK = (k0 + 32 <= K);

    // prefetch next K-chunk of B into cache while this chunk computes
    if (k0 + 32 < K) __builtin_prefetch(Bb + (long)(k0 + 32) * ldb + nblock, 0, 1);

    // ---- stage A tile (32x32) ----
    if (fullM && fullK && k4) {
      int r = t >> 3, c4 = (t & 7) << 2;
      const float4 v = *reinterpret_cast<const float4*>(Ab + (long)(mbase + r) * K + k0 + c4);
      As[r][c4 + 0] = (_Float16)v.x; As[r][c4 + 1] = (_Float16)v.y;
      As[r][c4 + 2] = (_Float16)v.z; As[r][c4 + 3] = (_Float16)v.w;
    } else {
      for (int i = t; i < 32 * 32; i += 256) {
        int r = i >> 5, c = i & 31;
        float v = 0.f;
        if (mbase + r < M && k0 + c < K) v = Ab[(long)(mbase + r) * K + (k0 + c)];
        As[r][c] = (_Float16)v;
      }
    }
    // ---- stage B tile (32x128) transposed into Bs[n][k] ----
    if (fullN && fullK) {
#pragma unroll
      for (int j = 0; j < 4; j++) {
        int i = t + j * 256;
        int r = i >> 5, c4 = (i & 31) << 2;
        const float4 v = *reinterpret_cast<const float4*>(Bb + (long)(k0 + r) * ldb + nblock + c4);
        Bs[c4 + 0][r] = (_Float16)v.x; Bs[c4 + 1][r] = (_Float16)v.y;
        Bs[c4 + 2][r] = (_Float16)v.z; Bs[c4 + 3][r] = (_Float16)v.w;
      }
    } else {
      for (int i = t; i < 32 * 128; i += 256) {
        int r = i >> 7, c = i & 127;   // r = k offset, c = n offset
        float v = 0.f;
        int gn = nblock + c;
        if (k0 + r < K && gn < N) v = Bb[(long)(k0 + r) * ldb + gn];
        Bs[c][r] = (_Float16)v;
      }
    }
    __syncthreads();

    v16h a0, a1, bf;
#pragma unroll
    for (int e = 0; e < 8; e++) {
      a0[e]     = As[row16][khalf * 8 + e];             // K = khalf*8 + e
      a0[e + 8] = As[row16][khalf * 8 + 16 + e];        // K = khalf*8 + 16 + e
      a1[e]     = As[16 + row16][khalf * 8 + e];
      a1[e + 8] = As[16 + row16][khalf * 8 + 16 + e];
    }
#pragma unroll
    for (int e = 0; e < 16; e++) {
      bf[e] = Bs[coln][khalf * 8 + (e & 7) + (e >> 3) * 16];  // K = khalf*8 + (e&7) + (e>>3)*16
    }

    acc0 = __builtin_amdgcn_wmma_f32_16x16x32_f16(
        false, a0, false, bf, (short)0, acc0, false, false);
    acc1 = __builtin_amdgcn_wmma_f32_16x16x32_f16(
        false, a1, false, bf, (short)0, acc1, false, false);
    __syncthreads();
  }

  const int nglob = nblock + coln;
  if (nglob < N) {
#pragma unroll
    for (int r = 0; r < 8; r++) {
      int m0 = mbase + r + (khalf ? 8 : 0);
      if (m0 < M) {
        float v = alpha * acc0[r];
        if (bias) v += bias[m0];
        Ob[(long)m0 * ldo + nglob] = d_act(v, act);
      }
      int m1 = m0 + 16;
      if (m1 < M) {
        float v = alpha * acc1[r];
        if (bias) v += bias[m1];
        Ob[(long)m1 * ldo + nglob] = d_act(v, act);
      }
    }
  }
}

// ---------------------------------------------------------------------------
// depthwise KxK conv2d, same padding, + bias (no activation)
// ---------------------------------------------------------------------------
__global__ void k_dw_conv2d(const float* __restrict__ x, long sX,
                            const float* __restrict__ w, const float* __restrict__ bias,
                            float* __restrict__ out, long sO,
                            int C, int Hh, int Wd, int Kk, int nb)
{
  long i = blockIdx.x * (long)blockDim.x + threadIdx.x;
  long tot = (long)nb * C * Hh * Wd;
  if (i >= tot) return;
  int wq = i % Wd; long r = i / Wd;
  int hq = r % Hh; r /= Hh;
  int c = r % C;  int b = r / C;
  int pad = Kk / 2;
  const float* xb = x + (long)b * sX + (long)c * Hh * Wd;
  float acc = bias[c];
  for (int ii = 0; ii < Kk; ii++) {
    int hy = hq + ii - pad; if (hy < 0 || hy >= Hh) continue;
    for (int jj = 0; jj < Kk; jj++) {
      int wx = wq + jj - pad; if (wx < 0 || wx >= Wd) continue;
      acc += w[((long)c * Kk + ii) * Kk + jj] * xb[(long)hy * Wd + wx];
    }
  }
  out[(long)b * sO + (long)c * Hh * Wd + (long)hq * Wd + wq] = acc;
}

// causal depthwise conv1d (D_CONV=4) along scan order + SiLU
__global__ void k_causal_conv1d_silu(const float* __restrict__ x, long sX,
                                     const float* __restrict__ w, const float* __restrict__ bias,
                                     float* __restrict__ out, long sO,
                                     int C, int Hh, int Wd, int dir, int nb)
{
  int L = Hh * Wd;
  long i = blockIdx.x * (long)blockDim.x + threadIdx.x;
  long tot = (long)nb * C * L;
  if (i >= tot) return;
  int tt = i % L; long r = i / L;
  int c = r % C; int b = r / C;
  const float* xb = x + (long)b * sX + (long)c * L;
  float acc = bias[c];
#pragma unroll
  for (int j = 0; j < 4; j++) {
    int tq = tt - 3 + j;
    if (tq >= 0) acc += w[c * 4 + j] * xb[perm_idx(dir, tq, Hh, Wd)];
  }
  out[(long)b * sO + (long)c * L + tt] = d_silu(acc);
}

// selective scan, one thread per (batch, channel); scatter-accumulate into
// ysum at original token position perm(t).
__global__ void k_selective_scan(const float* __restrict__ u, const float* __restrict__ dt, long sU,
                                 const float* __restrict__ Bm, const float* __restrict__ Cm, long sBC,
                                 const float* __restrict__ Alog, const float* __restrict__ Dv,
                                 float* __restrict__ ysum, long sY,
                                 int C, int Hh, int Wd, int dir)
{
  int c = threadIdx.x, b = blockIdx.x;
  int L = Hh * Wd;
  const float* ub = u  + (long)b * sU + (long)c * L;
  const float* db = dt + (long)b * sU + (long)c * L;
  const float* Bb = Bm + (long)b * sBC;
  const float* Cb = Cm + (long)b * sBC;
  float*       yb = ysum + (long)b * sY + (long)c * L;
  float A[kS], h[kS];
#pragma unroll
  for (int si = 0; si < kS; si++) { A[si] = -__expf(Alog[c * kS + si]); h[si] = 0.f; }
  float Dc = Dv[c];
  for (int t = 0; t < L; t++) {
    float dtv = db[t], uv = ub[t], y = 0.f;
#pragma unroll
    for (int si = 0; si < kS; si++) {
      h[si] = __expf(dtv * A[si]) * h[si] + dtv * Bb[(long)si * L + t] * uv;
      y += h[si] * Cb[(long)si * L + t];
    }
    yb[perm_idx(dir, t, Hh, Wd)] += y + uv * Dc;
  }
}

// RMSNorm over channel dim (columns of channel-major map), with pre-scale
__global__ void k_rmsnorm_cols(const float* __restrict__ x, const float* __restrict__ w,
                               float* __restrict__ out, int C, int L, float prescale)
{
  int n = blockIdx.x, b = blockIdx.y, c = threadIdx.x;
  __shared__ float red[128];
  float v = x[((long)b * C + c) * L + n] * prescale;
  red[c] = v * v;
  __syncthreads();
  for (int s2 = 64; s2 > 0; s2 >>= 1) { if (c < s2) red[c] += red[c + s2]; __syncthreads(); }
  float nrm = sqrtf(red[0]);
  out[((long)b * C + c) * L + n] = v / (nrm / sqrtf((float)C) + 1e-6f) * w[c];
}

// per-(b,d): mean and max over tokens -> ctx [B, 2D] (mean first)
__global__ void k_ctx_mean_max(const float* __restrict__ X, long sX, float* __restrict__ ctx,
                               int D, int L)
{
  int d = blockIdx.x, b = blockIdx.y, t = threadIdx.x;
  __shared__ float rs[256], rm[256];
  const float* row = X + (long)b * sX + (long)d * L;
  float s = 0.f, m = -3.4e38f;
  for (int i = t; i < L; i += 256) { float v = row[i]; s += v; m = fmaxf(m, v); }
  rs[t] = s; rm[t] = m;
  __syncthreads();
  for (int s2 = 128; s2 > 0; s2 >>= 1) {
    if (t < s2) { rs[t] += rs[t + s2]; rm[t] = fmaxf(rm[t], rm[t + s2]); }
    __syncthreads();
  }
  if (t == 0) { ctx[(long)b * 2 * D + d] = rs[0] / L; ctx[(long)b * 2 * D + D + d] = rm[0]; }
}

// small matvec: out[b,m] = act(W[m,:] . v[b,:] + bias[m] + addv[m])
__global__ void k_matvec(const float* __restrict__ Wm, const float* __restrict__ v, long sV,
                         const float* __restrict__ bias, const float* __restrict__ addv,
                         float* __restrict__ out, long sO, int M, int K, int act)
{
  int m = blockIdx.x, b = blockIdx.y, t = threadIdx.x;
  __shared__ float red[128];
  float s = 0.f;
  for (int k = t; k < K; k += 128) s += Wm[(long)m * K + k] * v[(long)b * sV + k];
  red[t] = s;
  __syncthreads();
  for (int s2 = 64; s2 > 0; s2 >>= 1) { if (t < s2) red[t] += red[t + s2]; __syncthreads(); }
  if (t == 0) {
    float r = red[0];
    if (bias) r += bias[m];
    if (addv) r += addv[m];
    out[(long)b * sO + m] = d_act(r, act);
  }
}

// softmax along length L rows (in place)
__global__ void k_softmax_rows(float* __restrict__ x, int L)
{
  long row = blockIdx.x;
  float* r = x + row * (long)L;
  __shared__ float red[256];
  int t = threadIdx.x;
  float m = -3.4e38f;
  for (int i = t; i < L; i += 256) m = fmaxf(m, r[i]);
  red[t] = m; __syncthreads();
  for (int s2 = 128; s2 > 0; s2 >>= 1) { if (t < s2) red[t] = fmaxf(red[t], red[t + s2]); __syncthreads(); }
  m = red[0]; __syncthreads();
  float sum = 0.f;
  for (int i = t; i < L; i += 256) { float e = __expf(r[i] - m); r[i] = e; sum += e; }
  red[t] = sum; __syncthreads();
  for (int s2 = 128; s2 > 0; s2 >>= 1) { if (t < s2) red[t] += red[t + s2]; __syncthreads(); }
  float inv = 1.f / red[0];
  __syncthreads();
  for (int i = t; i < L; i += 256) r[i] *= inv;
}

// He[b,d,e] = sum_n X_cm[b,d,n] * Asoft[b,e,n]
__global__ void k_he_reduce(const float* __restrict__ X, long sX,
                            const float* __restrict__ As, long sA,
                            float* __restrict__ He, long sH, int E, int L)
{
  int de = blockIdx.x, b = blockIdx.y, t = threadIdx.x;
  int d = de / E, e = de % E;
  __shared__ float red[256];
  const float* xr = X  + (long)b * sX + (long)d * L;
  const float* ar = As + (long)b * sA + (long)e * L;
  float s = 0.f;
  for (int i = t; i < L; i += 256) s += xr[i] * ar[i];
  red[t] = s; __syncthreads();
  for (int s2 = 128; s2 > 0; s2 >>= 1) { if (t < s2) red[t] += red[t + s2]; __syncthreads(); }
  if (t == 0) He[(long)b * sH + (long)d * E + e] = red[0];
}

// global average pool per (b,c)
__global__ void k_avgpool(const float* __restrict__ x, float* __restrict__ pool, int C, int L)
{
  int c = blockIdx.x, b = blockIdx.y, t = threadIdx.x;
  __shared__ float red[256];
  const float* r = x + ((long)b * C + c) * L;
  float s = 0.f;
  for (int i = t; i < L; i += 256) s += r[i];
  red[t] = s; __syncthreads();
  for (int s2 = 128; s2 > 0; s2 >>= 1) { if (t < s2) red[t] += red[t + s2]; __syncthreads(); }
  if (t == 0) pool[(long)b * C + c] = red[0] / L;
}

// ------------------------- elementwise helpers -----------------------------
__global__ void k_fill(float* __restrict__ p, long n, float v) {
  long i = blockIdx.x * (long)blockDim.x + threadIdx.x;
  if (i < n) p[i] = v;
}
__global__ void k_copy_rows(const float* __restrict__ src, long sS,
                            float* __restrict__ dst, long sD, int rows, int L, int nb) {
  long i = blockIdx.x * (long)blockDim.x + threadIdx.x;
  long tot = (long)nb * rows * L; if (i >= tot) return;
  int n = i % L; long r = i / L;
  int rr = r % rows; int b = r / rows;
  dst[(long)b * sD + (long)rr * L + n] = src[(long)b * sS + (long)rr * L + n];
}
__global__ void k_bcast_mul(const float* __restrict__ x, const float* __restrict__ z,
                            float* __restrict__ out, int C, int L, int nb) {
  long i = blockIdx.x * (long)blockDim.x + threadIdx.x;
  long tot = (long)nb * C * L; if (i >= tot) return;
  int n = i % L; long r = i / L; int b = (int)(r / C);
  out[i] = x[i] * z[(long)b * L + n];
}
__global__ void k_chan_gate(const float* __restrict__ x, const float* __restrict__ g,
                            float* __restrict__ out, int C, int L, int nb) {
  long i = blockIdx.x * (long)blockDim.x + threadIdx.x;
  long tot = (long)nb * C * L; if (i >= tot) return;
  long r = i / L; int c = r % C; int b = r / C;
  out[i] = x[i] * (1.f + g[(long)b * C + c]);
}
__global__ void k_add(const float* __restrict__ a, const float* __restrict__ bb,
                      float* __restrict__ out, long n) {
  long i = blockIdx.x * (long)blockDim.x + threadIdx.x;
  if (i < n) out[i] = a[i] + bb[i];
}
__global__ void k_mul_silu(const float* __restrict__ y, const float* __restrict__ z, long sZ,
                           float* __restrict__ out, int C, int L, int nb) {
  long i = blockIdx.x * (long)blockDim.x + threadIdx.x;
  long tot = (long)nb * C * L; if (i >= tot) return;
  int n = i % L; long r = i / L; int c = r % C; int b = r / C;
  out[i] = y[i] * d_silu(z[(long)b * sZ + (long)c * L + n]);
}
__global__ void k_axpy_gamma(const float* __restrict__ x, const float* __restrict__ t,
                             const float* __restrict__ gamma, float* __restrict__ out,
                             int C, int L, int nb) {
  long i = blockIdx.x * (long)blockDim.x + threadIdx.x;
  long tot = (long)nb * C * L; if (i >= tot) return;
  long r = i / L; int c = r % C;
  out[i] = x[i] + t[i] * gamma[c];
}
__global__ void k_final(const float* __restrict__ xin, const float* __restrict__ g,
                        const float* __restrict__ xm, float* __restrict__ out, long n) {
  long i = blockIdx.x * (long)blockDim.x + threadIdx.x;
  if (i < n) out[i] = xin[i] + d_sigmoid(g[i]) * xm[i];
}

// ---------------------------------------------------------------------------
// host-side helpers
// ---------------------------------------------------------------------------
static inline void gemm(hipStream_t s, const float* A, long sA,
                        const float* Bp, int ldb, long sB,
                        const float* bias, float* O, int ldo, long sO,
                        int M, int N, int K, int act, float alpha = 1.f, int batch = kB)
{
  dim3 g((N + 127) / 128, (M + 31) / 32, batch);
  wmma_gemm_bias_act<<<g, 256, 0, s>>>(A, sA, Bp, ldb, sB, bias, O, ldo, sO, M, N, K, alpha, act);
}
#define EWG(n) dim3((unsigned)(((n) + 255) / 256)), dim3(256)

struct C3AHP {
  const float *cv1w, *cv1b, *cv2w, *cv2b, *cv3w, *cv3b;
  const float *proto, *ctxw, *ctxb, *phpw, *phpb, *edgew, *edgeb, *nodew, *nodeb;
};
struct DSP { const float *dww, *dwb, *pww, *pwb; };

// c3ah branch: input y1 (rows 64..127 of T192), output -> Cat320 + rowoff*kHW
static void run_c3ah(hipStream_t s, const C3AHP& p, const float* yin, long sYin,
                     float* outCat, long sOutCat,
                     float* H0, float* H1, float* H2, float* H3, float* CatH,
                     float* ctx, float* protos, float* Lbuf, float* HeB, float* HeP)
{
  const long sH = (long)kCh * kHW;
  // a = silu(cv1 @ y1)
  gemm(s, p.cv1w, 0, yin, kHW, sYin, p.cv1b, H0, kHW, sH, kCh, kHW, kCh, ACT_SILU);
  // context (mean|max over tokens)
  k_ctx_mean_max<<<dim3(kCh, kB), 256, 0, s>>>(H0, sH, ctx, kCh, kHW);
  // protos = base + context_net(ctx)
  k_matvec<<<dim3(kE * kCh, kB), 128, 0, s>>>(p.ctxw, ctx, 2 * kCh, p.ctxb, p.proto,
                                              protos, kE * kCh, kE * kCh, 2 * kCh, ACT_NONE);
  // Xp = pre_head_proj(X)
  gemm(s, p.phpw, 0, H0, kHW, sH, p.phpb, H1, kHW, sH, kCh, kHW, kCh, ACT_NONE);
  // logits (head-mean already folded into alpha = 1/16)
  gemm(s, protos, kE * kCh, H1, kHW, sH, nullptr, Lbuf, kHW, (long)kE * kHW,
       kE, kHW, kCh, ACT_NONE, 1.f / 16.f);
  // softmax over nodes
  k_softmax_rows<<<dim3(kB * kE), 256, 0, s>>>(Lbuf, kHW);
  // He_raw^T[d,e]
  k_he_reduce<<<dim3(kCh * kE, kB), 256, 0, s>>>(H0, sH, Lbuf, (long)kE * kHW, HeB,
                                                 (long)kCh * kE, kE, kHW);
  // He = gelu(edge_proj(He_raw))   (channel-major [d_out, e])
  gemm(s, p.edgew, 0, HeB, kE, (long)kCh * kE, p.edgeb, HeP, kE, (long)kCh * kE,
       kCh, kE, kCh, ACT_GELU);
  // Xn_pre_cm = HeP @ Asoft
  gemm(s, HeP, (long)kCh * kE, Lbuf, kHW, (long)kE * kHW, nullptr, H2, kHW, sH,
       kCh, kHW, kE, ACT_NONE);
  // Xn = gelu(node_proj(Xn_pre)) + X
  gemm(s, p.nodew, 0, H2, kHW, sH, p.nodeb, H3, kHW, sH, kCh, kHW, kCh, ACT_GELU);
  k_add<<<EWG((long)kB * sH), 0, s>>>(H3, H0, H3, (long)kB * sH);
  // b = silu(cv2 @ y1)
  gemm(s, p.cv2w, 0, yin, kHW, sYin, p.cv2b, H1, kHW, sH, kCh, kHW, kCh, ACT_SILU);
  // concat [a', b] and cv3
  k_copy_rows<<<EWG((long)kB * sH), 0, s>>>(H3, sH, CatH, 2 * sH, kCh, kHW, kB);
  k_copy_rows<<<EWG((long)kB * sH), 0, s>>>(H1, sH, CatH + sH, 2 * sH, kCh, kHW, kB);
  gemm(s, p.cv3w, 0, CatH, kHW, 2 * sH, p.cv3b, outCat, kHW, sOutCat,
       kCh, kHW, 2 * kCh, ACT_SILU);
}

// ---------------------------------------------------------------------------
extern "C" void kernel_launch(void* const* d_in, const int* in_sizes, int n_in,
                              void* d_out, int out_size, void* d_ws, size_t ws_size,
                              hipStream_t stream)
{
  (void)in_sizes; (void)out_size; (void)ws_size;
  if (n_in < 90) return;
  hipStream_t s = stream;

  // ---- unpack parameters in setup_inputs() traversal order ----
  int idx = 0;
  auto nxt = [&]() -> const float* { return (const float*)d_in[idx++]; };
  const float *x1 = nxt(), *x2 = nxt();
  // cmca
  const float *c1w = nxt(), *c1b = nxt(), *c2w = nxt(), *c2b = nxt(), *c3w = nxt(), *c3b = nxt();
  const float *ci1w1 = nxt(), *ci1b1 = nxt(), *ci1w2 = nxt(), *ci1b2 = nxt();
  const float *ci2w1 = nxt(), *ci2b1 = nxt(), *ci2w2 = nxt(), *ci2b2 = nxt();
  const float *fusew = nxt(), *fuseb = nxt();
  const float *cv1w = nxt(), *cv1b = nxt(), *cv2w = nxt(), *cv2b = nxt();
  C3AHP br1, br2;
  auto loadC3 = [&](C3AHP& p) {
    p.cv1w = nxt(); p.cv1b = nxt(); p.cv2w = nxt(); p.cv2b = nxt(); p.cv3w = nxt(); p.cv3b = nxt();
    p.proto = nxt(); p.ctxw = nxt(); p.ctxb = nxt(); p.phpw = nxt(); p.phpb = nxt();
    p.edgew = nxt(); p.edgeb = nxt(); p.nodew = nxt(); p.nodeb = nxt();
  };
  loadC3(br1); loadC3(br2);
  // m0 (DSC3k)
  const float *m0cv1w = nxt(), *m0cv1b = nxt(), *m0cv2w = nxt(), *m0cv2b = nxt();
  const float *m0cv3w = nxt(), *m0cv3b = nxt();
  DSP ds[4];
  for (int i = 0; i < 4; i++) { ds[i].dww = nxt(); ds[i].dwb = nxt(); ds[i].pww = nxt(); ds[i].pwb = nxt(); }
  // mamba layer
  const float *normw = nxt(), *gamma = nxt(), *inpw = nxt();
  const float *c1dw = nxt(), *c1db = nxt(), *xprojw = nxt();
  const float *dtw = nxt(), *dtb = nxt(), *Alog = nxt(), *Dv = nxt();
  const float *onormw = nxt(), *oprojw = nxt();
  const float *op1w = nxt(), *op1b = nxt(), *op2w = nxt(), *op2b = nxt();

  // ---- workspace layout ----
  char* wsb = (char*)d_ws;
  size_t off = 0;
  auto alloc = [&](long nfloats) -> float* {
    float* p = (float*)(wsb + off);
    off += ((size_t)nfloats * sizeof(float) + 255) & ~(size_t)255;
    return p;
  };
  const long FM  = (long)kB * kC  * kHW;   // 128-ch map
  const long HM  = (long)kB * kCh * kHW;   // 64-ch map
  const long QM  = (long)kB * kQ  * kHW;   // 32-ch map
  float *F0 = alloc(FM), *F1 = alloc(FM), *F2 = alloc(FM), *F3 = alloc(FM), *F4 = alloc(FM), *F5 = alloc(FM);
  float *H0 = alloc(HM), *H1 = alloc(HM), *H2 = alloc(HM), *H3 = alloc(HM);
  float *Q0 = alloc(QM), *Q1 = alloc(QM), *Q2 = alloc(QM);
  float *CatA   = alloc((long)kB * 256 * kHW);
  float *Cat320 = alloc((long)kB * 320 * kHW);
  float *T192   = alloc((long)kB * 192 * kHW);
  float *CatH   = alloc((long)kB * 128 * kHW);
  float *zbuf   = alloc((long)kB * kHW);
  float *DBC    = alloc((long)kB * 24 * kHW);
  float *ctx    = alloc((long)kB * 2 * kCh);
  float *protos = alloc((long)kB * kE * kCh);
  float *Lbuf   = alloc((long)kB * kE * kHW);
  float *HeB    = alloc((long)kB * kCh * kE);
  float *HeP    = alloc((long)kB * kCh * kE);
  float *pool   = alloc((long)kB * kC);
  float *se16   = alloc((long)kB * 16);
  float *gate1  = alloc((long)kB * kC);
  float *gate2  = alloc((long)kB * kC);

  const long sF = (long)kC * kHW, sHalf = (long)kCh * kHW, sQm = (long)kQ * kHW;

  // =================== CMCA ===================
  gemm(s, c1w, 0, x1, kHW, sF, c1b, F0, kHW, sF, kC, kHW, kC, ACT_TANH);   // a1
  gemm(s, c2w, 0, x2, kHW, sF, c2b, F1, kHW, sF, kC, kHW, kC, ACT_TANH);   // a2
  k_copy_rows<<<EWG(FM), 0, s>>>(F0, sF, CatA, 2 * sF, kC, kHW, kB);
  k_copy_rows<<<EWG(FM), 0, s>>>(F1, sF, CatA + sF, 2 * sF, kC, kHW, kB);
  gemm(s, c3w, 0, CatA, kHW, 2 * sF, c3b, zbuf, kHW, (long)kHW, 1, kHW, 2 * kC, ACT_SIG); // z
  k_bcast_mul<<<EWG(FM), 0, s>>>(F0, zbuf, F2, kC, kHW, kB);   // m1
  k_bcast_mul<<<EWG(FM), 0, s>>>(F1, zbuf, F3, kC, kHW, kB);   // m2
  // gate1 = ci1(m2)
  k_avgpool<<<dim3(kC, kB), 256, 0, s>>>(F3, pool, kC, kHW);
  k_matvec<<<dim3(16, kB), 128, 0, s>>>(ci1w1, pool, kC, ci1b1, nullptr, se16, 16, 16, kC, ACT_GELU);
  k_matvec<<<dim3(kC, kB), 128, 0, s>>>(ci1w2, se16, 16, ci1b2, nullptr, gate1, kC, kC, 16, ACT_SIG);
  // gate2 = ci2(m1)
  k_avgpool<<<dim3(kC, kB), 256, 0, s>>>(F2, pool, kC, kHW);
  k_matvec<<<dim3(16, kB), 128, 0, s>>>(ci2w1, pool, kC, ci2b1, nullptr, se16, 16, 16, kC, ACT_GELU);
  k_matvec<<<dim3(kC, kB), 128, 0, s>>>(ci2w2, se16, 16, ci2b2, nullptr, gate2, kC, kC, 16, ACT_SIG);
  k_chan_gate<<<EWG(FM), 0, s>>>(F2, gate1, F0, kC, kHW, kB);  // o1
  k_chan_gate<<<EWG(FM), 0, s>>>(F3, gate2, F1, kC, kHW, kB);  // o2

  // =================== ChyperACE ===================
  k_copy_rows<<<EWG(FM), 0, s>>>(F0, sF, CatA, 2 * sF, kC, kHW, kB);
  k_copy_rows<<<EWG(FM), 0, s>>>(F1, sF, CatA + sF, 2 * sF, kC, kHW, kB);
  gemm(s, fusew, 0, CatA, kHW, 2 * sF, fuseb, F2, kHW, sF, kC, kHW, 2 * kC, ACT_SILU);      // x
  gemm(s, cv1w, 0, F2, kHW, sF, cv1b, T192, kHW, (long)192 * kHW, 192, kHW, kC, ACT_SILU);  // y0|y1|y2
  // y0 -> Cat320[0:64), y2 -> Cat320[128:192)
  k_copy_rows<<<EWG(HM), 0, s>>>(T192, (long)192 * kHW, Cat320, (long)320 * kHW, kCh, kHW, kB);
  k_copy_rows<<<EWG(HM), 0, s>>>(T192 + 2 * sHalf, (long)192 * kHW, Cat320 + 2 * sHalf,
                                 (long)320 * kHW, kCh, kHW, kB);
  const float* y1 = T192 + sHalf;
  const float* y2 = T192 + 2 * sHalf;
  // branches -> Cat320[64:128) and Cat320[256:320)
  run_c3ah(s, br1, y1, (long)192 * kHW, Cat320 + 1 * sHalf, (long)320 * kHW,
           H0, H1, H2, H3, CatH, ctx, protos, Lbuf, HeB, HeP);
  run_c3ah(s, br2, y1, (long)192 * kHW, Cat320 + 4 * sHalf, (long)320 * kHW,
           H0, H1, H2, H3, CatH, ctx, protos, Lbuf, HeB, HeP);
  // DSC3k(y2) -> Cat320[192:256)
  gemm(s, m0cv1w, 0, y2, kHW, (long)192 * kHW, m0cv1b, Q0, kHW, sQm, kQ, kHW, kCh, ACT_SILU);
  for (int blk = 0; blk < 2; blk++) {
    const DSP& d3 = ds[2 * blk + 0];   // k=3
    const DSP& d7 = ds[2 * blk + 1];   // k=7
    k_dw_conv2d<<<EWG(QM), 0, s>>>(Q0, sQm, d3.dww, d3.dwb, Q1, sQm, kQ, kH, kW, 3, kB);
    gemm(s, d3.pww, 0, Q1, kHW, sQm, d3.pwb, Q2, kHW, sQm, kQ, kHW, kQ, ACT_SILU);
    k_dw_conv2d<<<EWG(QM), 0, s>>>(Q2, sQm, d7.dww, d7.dwb, Q1, sQm, kQ, kH, kW, 7, kB);
    gemm(s, d7.pww, 0, Q1, kHW, sQm, d7.pwb, Q0, kHW, sQm, kQ, kHW, kQ, ACT_SILU);
  }
  gemm(s, m0cv2w, 0, y2, kHW, (long)192 * kHW, m0cv2b, Q1, kHW, sQm, kQ, kHW, kCh, ACT_SILU);
  k_copy_rows<<<EWG(QM), 0, s>>>(Q0, sQm, H0, sHalf, kQ, kHW, kB);
  k_copy_rows<<<EWG(QM), 0, s>>>(Q1, sQm, H0 + sQm, sHalf, kQ, kHW, kB);
  gemm(s, m0cv3w, 0, H0, kHW, sHalf, m0cv3b, Cat320 + 3 * sHalf, kHW, (long)320 * kHW,
       kCh, kHW, 2 * kQ, ACT_SILU);
  // cv2 over 5-way concat -> x_h
  gemm(s, cv2w, 0, Cat320, kHW, (long)320 * kHW, cv2b, F3, kHW, sF, kC, kHW, 320, ACT_SILU);

  // =================== Mamba block ===================
  k_rmsnorm_cols<<<dim3(kHW, kB), 128, 0, s>>>(F3, normw, F4, kC, kHW, 1.f);
  gemm(s, inpw, 0, F4, kHW, sF, nullptr, CatA, kHW, 2 * sF, 2 * kC, kHW, kC, ACT_NONE); // [x;z]
  k_fill<<<EWG(FM), 0, s>>>(F5, FM, 0.f);
  for (int dir = 0; dir < 4; dir++) {
    k_causal_conv1d_silu<<<EWG(FM), 0, s>>>(CatA, 2 * sF, c1dw, c1db, F0, sF, kC, kH, kW, dir, kB);
    gemm(s, xprojw, 0, F0, kHW, sF, nullptr, DBC, kHW, (long)24 * kHW,
         kDT + 2 * kS, kHW, kC, ACT_NONE);                       // [dt|B|C]
    gemm(s, dtw, 0, DBC, kHW, (long)24 * kHW, dtb, F1, kHW, sF, kC, kHW, kDT, ACT_SOFTPLUS);
    k_selective_scan<<<dim3(kB), kC, 0, s>>>(F0, F1, sF, DBC + (long)kDT * kHW,
                                             DBC + (long)(kDT + kS) * kHW, (long)24 * kHW,
                                             Alog + (long)dir * kC * kS, Dv, F5, sF,
                                             kC, kH, kW, dir);
  }
  k_rmsnorm_cols<<<dim3(kHW, kB), 128, 0, s>>>(F5, onormw, F0, kC, kHW, 0.25f);
  k_mul_silu<<<EWG(FM), 0, s>>>(F0, CatA + sF, 2 * sF, F1, kC, kHW, kB);
  gemm(s, oprojw, 0, F1, kHW, sF, nullptr, F2, kHW, sF, kC, kHW, kC, ACT_NONE);
  k_axpy_gamma<<<EWG(FM), 0, s>>>(F3, F2, gamma, F4, kC, kHW, kB);   // x_m

  // =================== gated outputs ===================
  gemm(s, op1w, 0, F3, kHW, sF, op1b, F0, kHW, sF, kC, kHW, kC, ACT_SILU);
  gemm(s, op2w, 0, F3, kHW, sF, op2b, F1, kHW, sF, kC, kHW, kC, ACT_SILU);
  float* out = (float*)d_out;
  k_final<<<EWG(FM), 0, s>>>(x1, F0, F4, out, FM);
  k_final<<<EWG(FM), 0, s>>>(x2, F1, F4, out + FM, FM);
}
